// EncoderBlock_62826781606021
// MI455X (gfx1250) — matrix operand order repaired and checked by reading, not compiled, not run on hardware
//
#include <hip/hip_runtime.h>

// ---------------------------------------------------------------------------
// Types
// ---------------------------------------------------------------------------
typedef __bf16  v16bf __attribute__((ext_vector_type(16)));
typedef float   v8f   __attribute__((ext_vector_type(8)));
typedef unsigned u32x4 __attribute__((ext_vector_type(4)));

union Frag {
    v16bf   v;
    unsigned u[8];
};

__device__ __forceinline__ unsigned short f2bfu(float f) {
    union { float f; unsigned u; } a; a.f = f;
    unsigned r = (a.u + 0x7fffu + ((a.u >> 16) & 1u)) >> 16;
    return (unsigned short)r;
}

// A-fragment K index pattern for 16-bit WMMA (contiguous: k = 16*half + 2*p)
__device__ __forceinline__ int a_kpat(int p, int half) {
    return 2 * p + 16 * half;
}

// ---------------------------------------------------------------------------
// fp32 -> bf16 conversion (weights)
// ---------------------------------------------------------------------------
__global__ __launch_bounds__(256) void cvt_bf16_kernel(const float* __restrict__ src,
                                                       unsigned short* __restrict__ dst,
                                                       int n) {
    int i = blockIdx.x * 256 + threadIdx.x;
    if (i < n) dst[i] = f2bfu(src[i]);
}

// ---------------------------------------------------------------------------
// LayerNorm (faithful: unbiased std, eps added to std), bf16 output
// one block per row, D = 1024
// ---------------------------------------------------------------------------
__global__ __launch_bounds__(256) void ln_kernel(const float* __restrict__ x,
                                                 const float* __restrict__ alpha,
                                                 const float* __restrict__ beta,
                                                 unsigned short* __restrict__ out) {
    const int D = 1024;
    int row = blockIdx.x;
    const float* xr = x + (size_t)row * D;
    float v[4];
    float s = 0.f, s2 = 0.f;
#pragma unroll
    for (int i = 0; i < 4; i++) {
        v[i] = xr[threadIdx.x + i * 256];
        s += v[i]; s2 += v[i] * v[i];
    }
#pragma unroll
    for (int off = 16; off > 0; off >>= 1) {
        s  += __shfl_xor(s, off);
        s2 += __shfl_xor(s2, off);
    }
    __shared__ float sh[2][8];
    int wave = threadIdx.x >> 5, lane = threadIdx.x & 31;
    if (lane == 0) { sh[0][wave] = s; sh[1][wave] = s2; }
    __syncthreads();
    if (wave == 0) {
        float a  = (lane < 8) ? sh[0][lane] : 0.f;
        float b2 = (lane < 8) ? sh[1][lane] : 0.f;
#pragma unroll
        for (int off = 4; off > 0; off >>= 1) {
            a  += __shfl_xor(a, off);
            b2 += __shfl_xor(b2, off);
        }
        if (lane == 0) { sh[0][0] = a; sh[1][0] = b2; }
    }
    __syncthreads();
    float S1 = sh[0][0], S2 = sh[1][0];
    float mean = S1 / (float)D;
    float var  = (S2 - (float)D * mean * mean) / (float)(D - 1);
    var = fmaxf(var, 0.f);
    float rdiv = 1.f / (sqrtf(var) + 1e-5f);
#pragma unroll
    for (int i = 0; i < 4; i++) {
        int c = threadIdx.x + i * 256;
        float o = alpha[c] * (v[i] - mean) * rdiv + beta[c];
        out[(size_t)row * D + c] = f2bfu(o);
    }
}

// ---------------------------------------------------------------------------
// Tiled bf16 WMMA GEMM:  C[M,N] = A[M,K](bf16) * B[K,N](bf16) + bias
// Compile-time variants (straight-line epilogue):
//   ACT:   0 none, 1 exact GELU
//   RESID: add fp32 residual
//   OUTF / OUTB: fp32 / bf16 outputs
// Block tile 128x128, 8 waves, each wave 64x32 (4x2 WMMA C tiles).
// Double-buffered LDS: global loads of tile i+1 overlap WMMA on tile i.
// ---------------------------------------------------------------------------
template <int ACT, bool RESID, bool OUTF, bool OUTB>
__global__ __launch_bounds__(256) void gemm_bf16_kernel(
    const unsigned short* __restrict__ A, const unsigned short* __restrict__ Bm,
    const float* __restrict__ bias, const float* __restrict__ resid,
    float* __restrict__ outF, unsigned short* __restrict__ outB,
    int M, int N, int K) {

    __shared__ __align__(16) unsigned short ldsA[2][128 * 32];   // [m][k], stride 32
    __shared__ __align__(16) unsigned short ldsB[2][128 * 40];   // [n][k], stride 40

    int tid  = threadIdx.x;
    int lane = tid & 31;
    int wave = tid >> 5;
    int wm = wave & 1;        // 2 M-slices of 64
    int wn = wave >> 1;       // 4 N-slices of 32
    int half = lane >> 4, l15 = lane & 15;
    int bM = blockIdx.y * 128, bN = blockIdx.x * 128;

    // per-thread staging coordinates
    int rA = tid >> 1, cA = (tid & 1) * 16;              // A: 16 elems of one row
    int kB = tid >> 3, nB = (tid & 7) * 16;              // B: 16 consecutive n, one k
    const unsigned short* Aptr = A + (size_t)(bM + rA) * K + cA;
    const unsigned short* Bptr = Bm + (size_t)kB * N + bN + nB;

    v8f acc[4][2] = {};

    // preload tile 0 -> LDS buffer 0
    u32x4 a0 = ((const u32x4*)Aptr)[0];
    u32x4 a1 = ((const u32x4*)Aptr)[1];
    u32x4 b0 = ((const u32x4*)Bptr)[0];
    u32x4 b1 = ((const u32x4*)Bptr)[1];
    {
        u32x4* dA = (u32x4*)(ldsA[0] + rA * 32 + cA);
        dA[0] = a0; dA[1] = a1;
        unsigned short tmp[16];
        *(u32x4*)&tmp[0] = b0;
        *(u32x4*)&tmp[8] = b1;
#pragma unroll
        for (int j = 0; j < 16; j++) ldsB[0][(nB + j) * 40 + kB] = tmp[j];
    }

    const int nk = K / 32;
    int cur = 0;
    for (int i = 0; i < nk; i++) {
        __syncthreads();
        // issue global loads for tile i+1 (hidden behind the WMMA phase)
        if (i + 1 < nk) {
            const unsigned short* Ap = Aptr + (i + 1) * 32;
            const unsigned short* Bp = Bptr + (size_t)(i + 1) * 32 * N;
            a0 = ((const u32x4*)Ap)[0];
            a1 = ((const u32x4*)Ap)[1];
            b0 = ((const u32x4*)Bp)[0];
            b1 = ((const u32x4*)Bp)[1];
        }

        // fragments from lds[cur] + 8 WMMAs
        Frag afr[4];
#pragma unroll
        for (int mt = 0; mt < 4; mt++) {
            const unsigned short* ap = ldsA[cur] + (wm * 64 + mt * 16 + l15) * 32;
#pragma unroll
            for (int p = 0; p < 8; p++)
                afr[mt].u[p] = *(const unsigned*)(ap + a_kpat(p, half));
        }
        Frag bfr[2];
#pragma unroll
        for (int nt = 0; nt < 2; nt++) {
            const unsigned short* bp =
                ldsB[cur] + (wn * 32 + nt * 16 + l15) * 40 + half * 16;
            *(u32x4*)&bfr[nt].u[0] = *(const u32x4*)bp;
            *(u32x4*)&bfr[nt].u[4] = *(const u32x4*)(bp + 8);
        }
#pragma unroll
        for (int mt = 0; mt < 4; mt++)
#pragma unroll
            for (int nt = 0; nt < 2; nt++)
                acc[mt][nt] = __builtin_amdgcn_wmma_f32_16x16x32_bf16(
                    false, afr[mt].v, false, bfr[nt].v, (short)0, acc[mt][nt],
                    false, false);

        // stage tile i+1 into the other buffer
        if (i + 1 < nk) {
            int nxt = cur ^ 1;
            u32x4* dA = (u32x4*)(ldsA[nxt] + rA * 32 + cA);
            dA[0] = a0; dA[1] = a1;
            unsigned short tmp[16];
            *(u32x4*)&tmp[0] = b0;
            *(u32x4*)&tmp[8] = b1;
#pragma unroll
            for (int j = 0; j < 16; j++) ldsB[nxt][(nB + j) * 40 + kB] = tmp[j];
        }
        cur ^= 1;
    }

    // Straight-line epilogue (all flags compile-time)
#pragma unroll
    for (int mt = 0; mt < 4; mt++) {
#pragma unroll
        for (int nt = 0; nt < 2; nt++) {
            int col = bN + wn * 32 + nt * 16 + l15;
            float bv = bias[col];
#pragma unroll
            for (int r = 0; r < 8; r++) {
                int row = bM + wm * 64 + mt * 16 + r + half * 8;
                float vv = acc[mt][nt][r] + bv;
                if (ACT == 1) vv = 0.5f * vv * (1.f + erff(vv * 0.70710678118f));
                size_t idx = (size_t)row * N + col;
                if (RESID) vv += resid[idx];
                if (OUTF) outF[idx] = vv;
                if (OUTB) outB[idx] = f2bfu(vv);
            }
        }
    }
}

// ---------------------------------------------------------------------------
// Flash-style attention (bf16 WMMA, fp32 accumulation).
// Softmax computed WITHOUT per-tile max subtraction (scores are O(1) here:
// exp stays far inside fp32 range; masked scores -> exp(-1e30) == 0), so the
// inner loop has no cross-lane reductions at all. The per-row denominator is
// accumulated as lane-private partial sums and reduced once at the end.
// Double-buffered K/V staging: one barrier per 32-key tile.
// grid = (B*H, S/128); block = 256 (8 waves); each wave: 16 q-rows x Dk=64.
// ---------------------------------------------------------------------------
__global__ __launch_bounds__(256) void attn_kernel(
    const unsigned short* __restrict__ qb, const unsigned short* __restrict__ kb,
    const unsigned short* __restrict__ vb, const int* __restrict__ mask,
    unsigned short* __restrict__ ctxb) {

    const int S = 2048, D = 1024, H = 16;
    int bh = blockIdx.x;
    int b = bh / H, h = bh % H;
    int q0 = blockIdx.y * 128;
    int tid = threadIdx.x, lane = tid & 31, wave = tid >> 5;
    int half = lane >> 4, l15 = lane & 15;

    __shared__ __align__(16) unsigned short ldsK[2][32 * 72];   // [key][d], stride 72
    __shared__ __align__(16) unsigned short ldsVt[2][64 * 40];  // [d][key], stride 40
    __shared__ __align__(16) unsigned short ldsP[8 * 16 * 36];  // per-wave P

    const int* maskb = mask + b * S;

    // Q fragments (2 K-chunks of 32 covering Dk=64), loaded once
    Frag qf[2];
    {
        int qrow = q0 + wave * 16 + l15;
        const unsigned short* qp = qb + ((size_t)(b * S + qrow)) * D + h * 64;
#pragma unroll
        for (int c = 0; c < 2; c++)
#pragma unroll
            for (int p = 0; p < 8; p++)
                qf[c].u[p] = *(const unsigned*)(qp + c * 32 + a_kpat(p, half));
    }

    v8f cacc[4] = {};
    float lsum[8];
#pragma unroll
    for (int r = 0; r < 8; r++) lsum[r] = 0.f;

    // K/V staging coordinates (per thread: one 16B chunk of K and of V)
    int srow = tid >> 3, sch = (tid & 7) * 8;
    size_t gbase = ((size_t)(b * S + srow)) * D + h * 64 + sch;

    // preload tile 0 -> buffer 0
    u32x4 kreg = *(const u32x4*)(kb + gbase);
    u32x4 vreg = *(const u32x4*)(vb + gbase);
    {
        *(u32x4*)(ldsK[0] + srow * 72 + sch) = kreg;
        unsigned short tmp[8];
        *(u32x4*)tmp = vreg;
#pragma unroll
        for (int j = 0; j < 8; j++) ldsVt[0][(sch + j) * 40 + srow] = tmp[j];
    }

    const int nt = S / 32;
    int cur = 0;
    for (int kt = 0; kt < nt; kt++) {
        __syncthreads();
        // global loads for tile kt+1 (overlap with compute)
        if (kt + 1 < nt) {
            size_t g = gbase + (size_t)(kt + 1) * 32 * D;
            kreg = *(const u32x4*)(kb + g);
            vreg = *(const u32x4*)(vb + g);
        }

        // scores: 16 q-rows x 32 keys  (two 16x16 C tiles)
        float scf[2][8];
#pragma unroll
        for (int st = 0; st < 2; st++) {
            v8f s = {};
#pragma unroll
            for (int c = 0; c < 2; c++) {
                Frag bf_;
                const unsigned short* kp =
                    ldsK[cur] + (st * 16 + l15) * 72 + c * 32 + half * 16;
                *(u32x4*)&bf_.u[0] = *(const u32x4*)kp;
                *(u32x4*)&bf_.u[4] = *(const u32x4*)(kp + 8);
                s = __builtin_amdgcn_wmma_f32_16x16x32_bf16(
                    false, qf[c].v, false, bf_.v, (short)0, s, false, false);
            }
            int mval = maskb[kt * 32 + st * 16 + l15];
#pragma unroll
            for (int r = 0; r < 8; r++)
                scf[st][r] = mval ? s[r] * 0.125f : -1e30f;
        }

        // probs (no reductions): exp, lane-private row-sum, P -> LDS (bf16)
        unsigned short* pb = ldsP + wave * (16 * 36);
#pragma unroll
        for (int r = 0; r < 8; r++) {
            float p0 = __expf(scf[0][r]);
            float p1 = __expf(scf[1][r]);
            lsum[r] += p0 + p1;
            int prow = r + half * 8;
            pb[prow * 36 + l15]      = f2bfu(p0);
            pb[prow * 36 + 16 + l15] = f2bfu(p1);
        }
        // same-wave LDS ops are in-order: no barrier needed for the P bounce
        Frag pf;
        const unsigned short* pp = pb + l15 * 36;
#pragma unroll
        for (int p = 0; p < 8; p++)
            pf.u[p] = *(const unsigned*)(pp + a_kpat(p, half));

        // ctx += P (16x32) x V (32x64), 4 WMMAs over Dk groups
#pragma unroll
        for (int dg = 0; dg < 4; dg++) {
            Frag vf;
            const unsigned short* vp =
                ldsVt[cur] + (dg * 16 + l15) * 40 + half * 16;
            *(u32x4*)&vf.u[0] = *(const u32x4*)vp;
            *(u32x4*)&vf.u[4] = *(const u32x4*)(vp + 8);
            cacc[dg] = __builtin_amdgcn_wmma_f32_16x16x32_bf16(
                false, pf.v, false, vf.v, (short)0, cacc[dg], false, false);
        }

        // stage tile kt+1 into the other buffer
        if (kt + 1 < nt) {
            int nxt = cur ^ 1;
            *(u32x4*)(ldsK[nxt] + srow * 72 + sch) = kreg;
            unsigned short tmp[8];
            *(u32x4*)tmp = vreg;
#pragma unroll
            for (int j = 0; j < 8; j++) ldsVt[nxt][(sch + j) * 40 + srow] = tmp[j];
        }
        cur ^= 1;
    }

    // one deferred denominator reduction, then normalize + store bf16 ctx
#pragma unroll
    for (int r = 0; r < 8; r++) {
        float rs = lsum[r];
        rs += __shfl_xor(rs, 1);
        rs += __shfl_xor(rs, 2);
        rs += __shfl_xor(rs, 4);
        rs += __shfl_xor(rs, 8);
        float inv = 1.f / rs;
        int orow = q0 + wave * 16 + r + half * 8;
        size_t base = ((size_t)(b * S + orow)) * D + h * 64;
#pragma unroll
        for (int dg = 0; dg < 4; dg++)
            ctxb[base + dg * 16 + l15] = f2bfu(cacc[dg][r] * inv);
    }
}

// ---------------------------------------------------------------------------
// Host-side orchestration
// ---------------------------------------------------------------------------
extern "C" void kernel_launch(void* const* d_in, const int* in_sizes, int n_in,
                              void* d_out, int out_size, void* d_ws, size_t ws_size,
                              hipStream_t stream) {
    const int Bb = 2, S = 2048, D = 1024, FF = 4096, H = 16;
    const int M = Bb * S;   // 4096 rows

    const float* x    = (const float*)d_in[0];
    const int*   mask = (const int*)  d_in[1];
    const float* wq = (const float*)d_in[2];  const float* bq = (const float*)d_in[3];
    const float* wk = (const float*)d_in[4];  const float* bk = (const float*)d_in[5];
    const float* wv = (const float*)d_in[6];  const float* bv = (const float*)d_in[7];
    const float* wo = (const float*)d_in[8];  const float* bo = (const float*)d_in[9];
    const float* w1 = (const float*)d_in[10]; const float* b1 = (const float*)d_in[11];
    const float* w2 = (const float*)d_in[12]; const float* b2 = (const float*)d_in[13];
    const float* ln1a = (const float*)d_in[14]; const float* ln1b = (const float*)d_in[15];
    const float* ln2a = (const float*)d_in[16]; const float* ln2b = (const float*)d_in[17];

    char* ws = (char*)d_ws;
    const size_t MB = 1024 * 1024;
    unsigned short* xn   = (unsigned short*)(ws + 0 * MB);   //  8 MB  (xn1, then xn2)
    unsigned short* qbuf = (unsigned short*)(ws + 8 * MB);   //  8 MB
    unsigned short* kbuf = (unsigned short*)(ws + 16 * MB);  //  8 MB
    unsigned short* vbuf = (unsigned short*)(ws + 24 * MB);  //  8 MB
    unsigned short* ctx  = (unsigned short*)(ws + 32 * MB);  //  8 MB
    unsigned short* hbuf = (unsigned short*)(ws + 8 * MB);   // 32 MB (aliases q/k/v/ctx)
    float*          x1   = (float*)(ws + 40 * MB);           // 16 MB
    unsigned short* wqb  = (unsigned short*)(ws + 56 * MB);  //  2 MB
    unsigned short* wkb  = (unsigned short*)(ws + 58 * MB);
    unsigned short* wvb  = (unsigned short*)(ws + 60 * MB);
    unsigned short* wob  = (unsigned short*)(ws + 62 * MB);
    unsigned short* w1b  = (unsigned short*)(ws + 64 * MB);  //  8 MB
    unsigned short* w2b  = (unsigned short*)(ws + 72 * MB);  //  8 MB

    auto cvt = [&](const float* s, unsigned short* dst, int n) {
        cvt_bf16_kernel<<<(n + 255) / 256, 256, 0, stream>>>(s, dst, n);
    };
    cvt(wq, wqb, D * D);
    cvt(wk, wkb, D * D);
    cvt(wv, wvb, D * D);
    cvt(wo, wob, D * D);
    cvt(w1, w1b, D * FF);
    cvt(w2, w2b, FF * D);

    // LN1 -> xn (bf16)
    ln_kernel<<<M, 256, 0, stream>>>(x, ln1a, ln1b, xn);

    // Q/K/V projections (bf16 out)
    dim3 gQKV(D / 128, M / 128);
    gemm_bf16_kernel<0, false, false, true><<<gQKV, 256, 0, stream>>>(
        xn, wqb, bq, nullptr, nullptr, qbuf, M, D, D);
    gemm_bf16_kernel<0, false, false, true><<<gQKV, 256, 0, stream>>>(
        xn, wkb, bk, nullptr, nullptr, kbuf, M, D, D);
    gemm_bf16_kernel<0, false, false, true><<<gQKV, 256, 0, stream>>>(
        xn, wvb, bv, nullptr, nullptr, vbuf, M, D, D);

    // Flash attention -> ctx (bf16)
    attn_kernel<<<dim3(Bb * H, S / 128), 256, 0, stream>>>(qbuf, kbuf, vbuf, mask, ctx);

    // Output projection + residual 1 -> x1 (fp32)
    gemm_bf16_kernel<0, true, true, false><<<gQKV, 256, 0, stream>>>(
        ctx, wob, bo, x, x1, nullptr, M, D, D);

    // LN2 -> xn (bf16, reuse buffer)
    ln_kernel<<<M, 256, 0, stream>>>(x1, ln2a, ln2b, xn);

    // FFN: GELU(xn @ w1 + b1) -> hbuf (bf16)
    gemm_bf16_kernel<1, false, false, true><<<dim3(FF / 128, M / 128), 256, 0, stream>>>(
        xn, w1b, b1, nullptr, nullptr, hbuf, M, FF, D);

    // hbuf @ w2 + b2 + x1 -> d_out (fp32)
    gemm_bf16_kernel<0, true, true, false><<<dim3(D / 128, M / 128), 256, 0, stream>>>(
        hbuf, w2b, b2, x1, (float*)d_out, nullptr, M, D, FF);
}
